// CondEquiUpdate_61211873903538
// MI455X (gfx1250) — matrix operand order repaired and checked by reading, not compiled, not run on hardware
//
#include <hip/hip_runtime.h>

typedef __bf16 v16bf __attribute__((ext_vector_type(16)));
typedef __bf16 v8bf  __attribute__((ext_vector_type(8)));
typedef __bf16 v4bf  __attribute__((ext_vector_type(4)));
typedef float  v8f   __attribute__((ext_vector_type(8)));

#define WAVES 4      // waves per block
#define MT    2      // 16-row sub-tiles per wave (B-fragment reuse factor)
#define TMW   32     // edges per wave-tile = MT*16
#define HD    128    // hidden
#define K1    320    // input channels
#define PAD1  328    // LDS row stride (halves) for staged input, bank-conflict free
#define PADX  136    // LDS row stride (halves) for x tile
#define NK1   10     // K1/32
#define NK2   4      // HD/32
#define NN    8      // HD/16 column tiles

__device__ __forceinline__ float silu_f(float x) { return x * (1.0f / (1.0f + __expf(-x))); }

// ---------------- FiLM prep: t = silu(time_emb) @ tW + tb ; split shift/scale
__global__ void film_kernel(const float* __restrict__ time_emb,
                            const float* __restrict__ tW,
                            const float* __restrict__ tb,
                            float* __restrict__ ws_shift,
                            float* __restrict__ ws_scale) {
  int j = threadIdx.x;              // 0..255
  float s = tb[j];
  for (int i = 0; i < 128; ++i) {
    float te = time_emb[i];
    s += silu_f(te) * tW[i * 256 + j];
  }
  if (j < 128) ws_shift[j] = s;
  else         ws_scale[j - 128] = s;
}

// ---------------- Pack iW (320x128) and c1W (128x128) into bf16 WMMA B-fragment order.
// B fragment (16-bit, 32x16): elem i<8 -> K=8*(lane/16)+i, elem i>=8 -> K=16+8*(lane/16)+(i-8); N = lane%16.
// Packed index: ((kk*NN + nn)*32 + lane)*16 + i
__global__ void pack_kernel(const float* __restrict__ iW,
                            const float* __restrict__ c1W,
                            __bf16* __restrict__ wiW,
                            __bf16* __restrict__ wc1) {
  int o = blockIdx.x * 256 + threadIdx.x;
  const int NIW = NK1 * NN * 32 * 16;   // 40960
  const int NC1 = NK2 * NN * 32 * 16;   // 16384
  if (o < NIW) {
    int i    = o & 15;
    int lane = (o >> 4) & 31;
    int nn   = (o >> 9) & 7;
    int kk   = o >> 12;
    int k = kk * 32 + ((lane >> 4) << 3) + (i & 7) + ((i >> 3) << 4);
    int n = nn * 16 + (lane & 15);
    wiW[o] = (__bf16)iW[k * HD + n];
  } else if (o < NIW + NC1) {
    int o2 = o - NIW;
    int i    = o2 & 15;
    int lane = (o2 >> 4) & 31;
    int nn   = (o2 >> 9) & 7;
    int kk   = o2 >> 12;
    int k = kk * 32 + ((lane >> 4) << 3) + (i & 7) + ((i >> 3) << 4);
    int n = nn * 16 + (lane & 15);
    wc1[o2] = (__bf16)c1W[k * HD + n];
  }
}

// ---------------- out = pos (output accumulates scatter-adds on top)
__global__ void init_kernel(const float* __restrict__ pos, float* __restrict__ out, int n) {
  int i = blockIdx.x * 256 + threadIdx.x;
  if (i < n) out[i] = pos[i];
}

// ---------------- Main fused edge kernel: one wave handles a 32-edge tile (2 WMMA row-tiles).
__global__ __launch_bounds__(WAVES * 32)
void main_kernel(const float* __restrict__ h,
                 const float* __restrict__ pos,
                 const float* __restrict__ ea,
                 const float* __restrict__ dist,
                 const long long* __restrict__ eidx,
                 const float* __restrict__ ib,
                 const float* __restrict__ c1b,
                 const float* __restrict__ c2w,
                 const float* __restrict__ cns,
                 const float* __restrict__ ws_shift,
                 const float* __restrict__ ws_scale,
                 const v16bf* __restrict__ wiW,
                 const v16bf* __restrict__ wc1,
                 float* __restrict__ out,
                 long long E, long long ntiles) {
  __shared__ __bf16 s_in[WAVES][TMW][PAD1];
  __shared__ float  s_cd[WAVES][TMW][4];
  __shared__ int    s_row[WAVES][TMW];
  __shared__ int    s_col[WAVES][TMW];

  const int wave = threadIdx.x >> 5;
  const int lane = threadIdx.x & 31;
  const long long tile = (long long)blockIdx.x * WAVES + wave;
  if (tile >= ntiles) return;
  const long long edge0 = tile * TMW;
  const int lrow = lane & 15;     // M row / N col within 16-lane group
  const int hgrp = lane >> 4;     // lane half

  // Per-lane fixed column (N) preloads: each lane owns columns nn*16 + lrow
  float scalev[NN], shiftv[NN], ibv[NN], c1bv[NN], c2wv[NN];
#pragma unroll
  for (int nn = 0; nn < NN; ++nn) {
    int n = nn * 16 + lrow;
    scalev[nn] = ws_scale[n];
    shiftv[nn] = ws_shift[n];
    ibv[nn]    = ib[n];
    c1bv[nn]   = c1b[n];
    c2wv[nn]   = c2w[n];
  }
  const float cn = cns[0];

  // ---- stage edge indices + normalized coord diffs (one edge per lane)
  {
    long long r = eidx[edge0 + lane];
    long long c = eidx[E + edge0 + lane];
    s_row[wave][lane] = (int)r;
    s_col[wave][lane] = (int)c;
    float dx = pos[r * 3 + 0] - pos[c * 3 + 0];
    float dy = pos[r * 3 + 1] - pos[c * 3 + 1];
    float dz = pos[r * 3 + 2] - pos[c * 3 + 2];
    float nrm = sqrtf(dx * dx + dy * dy + dz * dz);
    float sc = cn / fmaxf(nrm, 1e-8f);
    s_cd[wave][lane][0] = dx * sc;
    s_cd[wave][lane][1] = dy * sc;
    s_cd[wave][lane][2] = dz * sc;
  }
  asm volatile("" ::: "memory");  // same-wave DS ops are in-order; fence the compiler

  // ---- gather h[row], h[col], edge_attr, dist -> bf16 LDS tile [32][320]
  for (int e = 0; e < TMW; ++e) {
    long long r = s_row[wave][e];
    long long c = s_col[wave][e];
    float4 hv = ((const float4*)(h + r * HD))[lane];   // 32 lanes x 4 = 128 floats
    v4bf b0; b0.x = (__bf16)hv.x; b0.y = (__bf16)hv.y; b0.z = (__bf16)hv.z; b0.w = (__bf16)hv.w;
    *(v4bf*)&s_in[wave][e][lane * 4] = b0;
    float4 hc = ((const float4*)(h + c * HD))[lane];
    v4bf b1; b1.x = (__bf16)hc.x; b1.y = (__bf16)hc.y; b1.z = (__bf16)hc.z; b1.w = (__bf16)hc.w;
    *(v4bf*)&s_in[wave][e][HD + lane * 4] = b1;
    s_in[wave][e][256 + lane] = (__bf16)ea[(edge0 + e) * 32 + lane];
    s_in[wave][e][288 + lane] = (__bf16)dist[(edge0 + e) * 32 + lane];
  }
  asm volatile("" ::: "memory");

  // ---- GEMM1: x[32x128] = h_input[32x320] @ iW, f32 accum via bf16 WMMA.
  // Two 16-row sub-tiles share every B fragment; all 8 B frags of a k-step are
  // preloaded so load latency is pipelined against the 16-WMMA burst.
  v8f acc0[NN], acc1[NN];
  const v8f vzero = {0.f, 0.f, 0.f, 0.f, 0.f, 0.f, 0.f, 0.f};
#pragma unroll
  for (int nn = 0; nn < NN; ++nn) { acc0[nn] = vzero; acc1[nn] = vzero; }

  const __bf16* srow0 = &s_in[wave][lrow][0];
  const __bf16* srow1 = &s_in[wave][16 + lrow][0];
#pragma unroll
  for (int kk = 0; kk < NK1; ++kk) {
    // A frag (16x32 bf16): elems 0-7 = K kb..kb+7, elems 8-15 = K kb+16..kb+23, kb = 8*hgrp
    const __bf16* p0 = srow0 + kk * 32 + hgrp * 8;
    v8bf lo0 = *(const v8bf*)(p0);
    v8bf hi0 = *(const v8bf*)(p0 + 16);
    v16bf a0 = __builtin_shufflevector(lo0, hi0, 0, 1, 2, 3, 4, 5, 6, 7,
                                       8, 9, 10, 11, 12, 13, 14, 15);
    const __bf16* p1 = srow1 + kk * 32 + hgrp * 8;
    v8bf lo1 = *(const v8bf*)(p1);
    v8bf hi1 = *(const v8bf*)(p1 + 16);
    v16bf a1 = __builtin_shufflevector(lo1, hi1, 0, 1, 2, 3, 4, 5, 6, 7,
                                       8, 9, 10, 11, 12, 13, 14, 15);
    v16bf bfr[NN];
#pragma unroll
    for (int nn = 0; nn < NN; ++nn)
      bfr[nn] = wiW[(kk * NN + nn) * 32 + lane];
#pragma unroll
    for (int nn = 0; nn < NN; ++nn) {
      acc0[nn] = __builtin_amdgcn_wmma_f32_16x16x32_bf16(
          false, a0, false, bfr[nn], (short)0, acc0[nn], false, false);
      acc1[nn] = __builtin_amdgcn_wmma_f32_16x16x32_bf16(
          false, a1, false, bfr[nn], (short)0, acc1[nn], false, false);
    }
  }

  // ---- bias + LayerNorm(eps=1e-6, no affine) + FiLM, in registers, both sub-tiles.
  // C layout: (vgpr r, lane) -> row M = r + 8*hgrp (+16*t), col N = lane%16 (+16*nn)
#pragma unroll
  for (int t = 0; t < MT; ++t) {
    v8f* acct = t ? acc1 : acc0;
#pragma unroll
    for (int r = 0; r < 8; ++r) {
      float v[NN];
      float s1 = 0.f, s2 = 0.f;
#pragma unroll
      for (int nn = 0; nn < NN; ++nn) {
        float tv = acct[nn][r] + ibv[nn];
        v[nn] = tv;
        s1 += tv;
        s2 += tv * tv;
      }
#pragma unroll
      for (int m = 8; m >= 1; m >>= 1) {   // reduce over 16 cols (stays within lane half)
        s1 += __shfl_xor(s1, m, 32);
        s2 += __shfl_xor(s2, m, 32);
      }
      float mu  = s1 * (1.0f / 128.0f);
      float var = s2 * (1.0f / 128.0f) - mu * mu;
      float rs  = rsqrtf(var + 1e-6f);
#pragma unroll
      for (int nn = 0; nn < NN; ++nn) {
        float xn = (v[nn] - mu) * rs;
        acct[nn][r] = xn * (1.0f + scalev[nn]) + shiftv[nn];
      }
    }
  }

  // ---- restage x as bf16 [32][128] (stride PADX) for GEMM2 A-frags
  __bf16* sx = (__bf16*)&s_in[wave][0][0];
#pragma unroll
  for (int t = 0; t < MT; ++t) {
    v8f* acct = t ? acc1 : acc0;
#pragma unroll
    for (int r = 0; r < 8; ++r) {
      int row = t * 16 + r + 8 * hgrp;
#pragma unroll
      for (int nn = 0; nn < NN; ++nn)
        sx[row * PADX + nn * 16 + lrow] = (__bf16)acct[nn][r];
    }
  }
  asm volatile("" ::: "memory");

  // ---- GEMM2: y[32x128] = x @ c1W
  v8f y0[NN], y1[NN];
#pragma unroll
  for (int nn = 0; nn < NN; ++nn) { y0[nn] = vzero; y1[nn] = vzero; }
  const __bf16* sxr0 = sx + lrow * PADX;
  const __bf16* sxr1 = sx + (16 + lrow) * PADX;
#pragma unroll
  for (int kk = 0; kk < NK2; ++kk) {
    const __bf16* p0 = sxr0 + kk * 32 + hgrp * 8;
    v8bf lo0 = *(const v8bf*)(p0);
    v8bf hi0 = *(const v8bf*)(p0 + 16);
    v16bf a0 = __builtin_shufflevector(lo0, hi0, 0, 1, 2, 3, 4, 5, 6, 7,
                                       8, 9, 10, 11, 12, 13, 14, 15);
    const __bf16* p1 = sxr1 + kk * 32 + hgrp * 8;
    v8bf lo1 = *(const v8bf*)(p1);
    v8bf hi1 = *(const v8bf*)(p1 + 16);
    v16bf a1 = __builtin_shufflevector(lo1, hi1, 0, 1, 2, 3, 4, 5, 6, 7,
                                       8, 9, 10, 11, 12, 13, 14, 15);
    v16bf bfr[NN];
#pragma unroll
    for (int nn = 0; nn < NN; ++nn)
      bfr[nn] = wc1[(kk * NN + nn) * 32 + lane];
#pragma unroll
    for (int nn = 0; nn < NN; ++nn) {
      y0[nn] = __builtin_amdgcn_wmma_f32_16x16x32_bf16(
          false, a0, false, bfr[nn], (short)0, y0[nn], false, false);
      y1[nn] = __builtin_amdgcn_wmma_f32_16x16x32_bf16(
          false, a1, false, bfr[nn], (short)0, y1[nn], false, false);
    }
  }

  // ---- inv = tanh( silu(y + c1b) . c2W ) per row, then scatter both sub-tiles
  float myinv[MT];
#pragma unroll
  for (int t = 0; t < MT; ++t) {
    v8f* yt = t ? y1 : y0;
    myinv[t] = 0.f;
#pragma unroll
    for (int r = 0; r < 8; ++r) {
      float s = 0.f;
#pragma unroll
      for (int nn = 0; nn < NN; ++nn) {
        float yv = yt[nn][r] + c1bv[nn];
        s += silu_f(yv) * c2wv[nn];
      }
#pragma unroll
      for (int m = 8; m >= 1; m >>= 1) s += __shfl_xor(s, m, 32);
      float inv = tanhf(s);
      if (lrow == r) myinv[t] = inv;  // lane j keeps row j (half 0), lane 16+j keeps row 8+j
    }
  }

  // ---- scatter: trans = coord_diff * inv, atomic segment-sum into out
  if (lrow < 8) {
#pragma unroll
    for (int t = 0; t < MT; ++t) {
      int e = t * 16 + lrow + 8 * hgrp;        // edge within tile
      int nid = s_row[wave][e];
      float cdx = s_cd[wave][e][0] * myinv[t];
      float cdy = s_cd[wave][e][1] * myinv[t];
      float cdz = s_cd[wave][e][2] * myinv[t];
      __hip_atomic_fetch_add(&out[nid * 3 + 0], cdx, __ATOMIC_RELAXED, __HIP_MEMORY_SCOPE_AGENT);
      __hip_atomic_fetch_add(&out[nid * 3 + 1], cdy, __ATOMIC_RELAXED, __HIP_MEMORY_SCOPE_AGENT);
      __hip_atomic_fetch_add(&out[nid * 3 + 2], cdz, __ATOMIC_RELAXED, __HIP_MEMORY_SCOPE_AGENT);
    }
  }
}

extern "C" void kernel_launch(void* const* d_in, const int* in_sizes, int n_in,
                              void* d_out, int out_size, void* d_ws, size_t ws_size,
                              hipStream_t stream) {
  const float* h        = (const float*)d_in[0];
  const float* pos      = (const float*)d_in[1];
  const float* ea       = (const float*)d_in[2];
  const float* dist     = (const float*)d_in[3];
  const float* time_emb = (const float*)d_in[4];
  const float* tW       = (const float*)d_in[5];
  const float* tb       = (const float*)d_in[6];
  const float* iW       = (const float*)d_in[7];
  const float* ib       = (const float*)d_in[8];
  const float* c1W      = (const float*)d_in[9];
  const float* c1b      = (const float*)d_in[10];
  const float* c2w      = (const float*)d_in[11];
  const float* cns      = (const float*)d_in[12];
  const long long* eidx = (const long long*)d_in[13];
  float* out = (float*)d_out;

  const long long E = (long long)in_sizes[13] / 2;   // 640000
  const int N3 = in_sizes[1];                        // N*3

  // Workspace layout
  float* ws_shift = (float*)d_ws;                    // 128 f32
  float* ws_scale = ws_shift + 128;                  // 128 f32
  __bf16* wiW = (__bf16*)(ws_scale + 128);           // 40960 bf16 (32B aligned)
  __bf16* wc1 = wiW + NK1 * NN * 32 * 16;            // 16384 bf16

  film_kernel<<<1, 256, 0, stream>>>(time_emb, tW, tb, ws_shift, ws_scale);

  const int npack = (NK1 + NK2) * NN * 32 * 16;      // 57344
  pack_kernel<<<(npack + 255) / 256, 256, 0, stream>>>(iW, c1W, wiW, wc1);

  init_kernel<<<(N3 + 255) / 256, 256, 0, stream>>>(pos, out, N3);

  const long long ntiles = E / TMW;                  // full 32-edge tiles (E divisible by 32)
  const int nblocks = (int)((ntiles + WAVES - 1) / WAVES);
  main_kernel<<<nblocks, WAVES * 32, 0, stream>>>(
      h, pos, ea, dist, eidx, ib, c1b, c2w, cns,
      ws_shift, ws_scale, (const v16bf*)wiW, (const v16bf*)wc1, out, E, ntiles);
}
